// PureMamba_18622978195558
// MI455X (gfx1250) — matrix operand order repaired and checked by reading, not compiled, not run on hardware
//
#include <hip/hip_runtime.h>

// ---------------------------------------------------------------------------
// Mamba forward on MI455X (gfx1250, wave32).
//  - GEMMs via v_wmma_f32_16x16x32_f16, one 16x32 D-slab per wave (A reused
//    across two N-tiles), fragments loaded as global_load_b128 pairs from
//    pre-converted f16 weights (converted once per launch).
//  - selective scan via chunked 2-pass scan (chunk=128, 16 chunks) for
//    65K-lane parallelism in the transcendental-heavy passes.
//  - wave-wide shfl_xor reduction over the 64 complex states per channel.
// ---------------------------------------------------------------------------

typedef __attribute__((ext_vector_type(16))) _Float16 v16h;
typedef __attribute__((ext_vector_type(8)))  _Float16 v8h;
typedef __attribute__((ext_vector_type(8)))  float    v8f;

#define LSEQ   2048
#define DMODEL 64
#define DINNER 128
#define DSTATE 64
#define NXROW  260   // DT_RANK(4) + 4*DSTATE
#define NLAYER 16
#define CHUNK  128
#define NCH    (LSEQ / CHUNK)

// ------------------------------ init --------------------------------------
__global__ void k_init(const float* __restrict__ x,
                       float* __restrict__ hidden, float* __restrict__ residual) {
  int i = blockIdx.x * blockDim.x + threadIdx.x;
  hidden[i]   = x[i];
  residual[i] = 0.f;
}

// ----------------------- f32 -> f16 weight convert -------------------------
__global__ void k_cvt(const float* __restrict__ src, _Float16* __restrict__ dst) {
  int i = blockIdx.x * blockDim.x + threadIdx.x;
  dst[i] = (_Float16)src[i];
}

// --------------------- residual add + RMSNorm -> f16 -----------------------
__global__ void k_norm(const float* __restrict__ hidden, float* __restrict__ residual,
                       const float* __restrict__ nw, _Float16* __restrict__ h16) {
  int l = blockIdx.x;
  int i = threadIdx.x;                       // 0..63
  float v = hidden[l * DMODEL + i] + residual[l * DMODEL + i];
  residual[l * DMODEL + i] = v;
  float ss = v * v;
  #pragma unroll
  for (int off = 16; off; off >>= 1) ss += __shfl_xor(ss, off);
  __shared__ float sbuf[2];
  if ((threadIdx.x & 31) == 0) sbuf[threadIdx.x >> 5] = ss;
  __syncthreads();
  float scale = rsqrtf((sbuf[0] + sbuf[1]) * (1.f / DMODEL) + 1e-5f);
  h16[l * DMODEL + i] = (_Float16)(v * scale * nw[i]);
}

// ------------------------- WMMA GEMM: D = A * W^T --------------------------
// A: (M x K) f16 row-major, W: (N x K) f16 row-major, D: (M x N) f32.
// M % 16 == 0, K % 32 == 0 assumed; N may be ragged (clamped loads, the
// garbage accumulates only into D-columns n >= N which are never stored).
// One wave computes a 16x32 slab (two 16x16 tiles, A fragment reused).
__global__ void k_gemm(const _Float16* __restrict__ A, int lda,
                       const _Float16* __restrict__ W, int ldw,
                       float* __restrict__ Dm, int ldd,
                       int M, int N, int K, int nTilePairs) {
  int gwave = (blockIdx.x * blockDim.x + threadIdx.x) >> 5;
  int lane  = threadIdx.x & 31;
  int tm = gwave / nTilePairs;
  int tp = gwave % nTilePairs;
  if (tm * 16 >= M) return;                  // wave-uniform: EXEC stays all-1
  int row0 = tm * 16;
  int r    = row0 + (lane & 15);
  int n0   = tp * 32 + (lane & 15);
  int n1   = n0 + 16;
  // A layout: lanes<16 hold K 0-15 of row r; lanes>=16 hold K 16-31
  int kb   = (lane & 16) ? 16 : 0;
  // B layout: lanes<16 hold K 0-15 of column n; lanes>=16 hold K 16-31
  int kbb  = (lane & 16) ? 16 : 0;
  const _Float16* arow  = A + r * lda + kb;
  const _Float16* wrow0 = W + ((n0 < N) ? n0 : (N - 1)) * ldw + kbb;
  const _Float16* wrow1 = W + ((n1 < N) ? n1 : (N - 1)) * ldw + kbb;
  __builtin_prefetch(wrow0, 0, 3);           // global_prefetch_b8

  v8f acc0 = {}, acc1 = {};
  for (int k0 = 0; k0 < K; k0 += 32) {
    v8h alo  = *(const v8h*)(arow + k0);
    v8h ahi  = *(const v8h*)(arow + k0 + 8);
    v8h b0lo = *(const v8h*)(wrow0 + k0);
    v8h b0hi = *(const v8h*)(wrow0 + k0 + 8);
    v8h b1lo = *(const v8h*)(wrow1 + k0);
    v8h b1hi = *(const v8h*)(wrow1 + k0 + 8);
    v16h a  = __builtin_shufflevector(alo,  ahi,  0,1,2,3,4,5,6,7,8,9,10,11,12,13,14,15);
    v16h b0 = __builtin_shufflevector(b0lo, b0hi, 0,1,2,3,4,5,6,7,8,9,10,11,12,13,14,15);
    v16h b1 = __builtin_shufflevector(b1lo, b1hi, 0,1,2,3,4,5,6,7,8,9,10,11,12,13,14,15);
    acc0 = __builtin_amdgcn_wmma_f32_16x16x32_f16(false, a, false, b0,
                                                  (short)0, acc0, false, false);
    acc1 = __builtin_amdgcn_wmma_f32_16x16x32_f16(false, a, false, b1,
                                                  (short)0, acc1, false, false);
  }
  // D layout: VGPR r8 -> M = row0 + r8 (+8 for lanes>=16), N = n0/n1
  int mb = (lane & 16) ? 8 : 0;
  if (n0 < N) {
    #pragma unroll
    for (int r8 = 0; r8 < 8; ++r8) Dm[(row0 + mb + r8) * ldd + n0] = acc0[r8];
  }
  if (n1 < N) {
    #pragma unroll
    for (int r8 = 0; r8 < 8; ++r8) Dm[(row0 + mb + r8) * ldd + n1] = acc1[r8];
  }
}

// ------------------- causal depthwise conv(4) + SiLU -----------------------
__global__ void k_conv(const float* __restrict__ xz, const float* __restrict__ cw,
                       const float* __restrict__ cb,
                       float* __restrict__ xc32, _Float16* __restrict__ xc16) {
  int idx = blockIdx.x * blockDim.x + threadIdx.x;   // l*128 + d
  int l = idx >> 7, d = idx & 127;
  float acc = cb[d];
  #pragma unroll
  for (int k = 0; k < 4; ++k) {
    int ls = l - 3 + k;
    float xv = (ls >= 0) ? xz[ls * (2 * DINNER) + d] : 0.f;
    acc += xv * cw[d * 4 + k];
  }
  float s = acc / (1.f + __expf(-acc));              // silu
  xc32[idx] = s;
  xc16[idx] = (_Float16)s;
}

// --------------------- dt = softplus(dtr @ dpw^T + bias) -------------------
__global__ void k_dt(const float* __restrict__ xdbl, const float* __restrict__ dpw,
                     const float* __restrict__ dbias, float* __restrict__ dtb) {
  int idx = blockIdx.x * blockDim.x + threadIdx.x;   // l*128 + d
  int l = idx >> 7, d = idx & 127;
  float a = dbias[d];
  #pragma unroll
  for (int r = 0; r < 4; ++r) a += xdbl[l * NXROW + r] * dpw[d * 4 + r];
  dtb[idx] = (a > 20.f) ? a : log1pf(__expf(a));
}

// -------------- scan pass 1: per-chunk decay product & partial h -----------
// one wave per (chunk, channel); each lane owns states s and s+32.
__global__ void k_scan1(const float* __restrict__ dtb, const float* __restrict__ xc,
                        const float* __restrict__ xdbl,
                        const float* __restrict__ alog, const float* __restrict__ aimg,
                        float* __restrict__ chp, float* __restrict__ chh) {
  int gwave = (blockIdx.x * blockDim.x + threadIdx.x) >> 5;
  int lane  = threadIdx.x & 31;
  int c = gwave >> 7, d = gwave & 127;
  float ar[2], ai[2], hr[2] = {0, 0}, hi[2] = {0, 0}, pr[2] = {1, 1}, pi[2] = {0, 0};
  #pragma unroll
  for (int q = 0; q < 2; ++q) {
    int s = lane + q * 32;
    ar[q] = -__expf(alog[d * DSTATE + s]);
    ai[q] = aimg[d * DSTATE + s];
  }
  int tbeg = c * CHUNK;
  for (int t = tbeg; t < tbeg + CHUNK; ++t) {
    float dv = dtb[t * DINNER + d];
    float u  = dv * xc[t * DINNER + d];
    const float* row = xdbl + t * NXROW;
    #pragma unroll
    for (int q = 0; q < 2; ++q) {
      int s = lane + q * 32;
      float ea  = __expf(dv * ar[q]);
      float th  = dv * ai[q];
      float atr = ea * __cosf(th), ati = ea * __sinf(th);
      float br = u * row[4 + s], bi = u * row[68 + s];
      float nr = atr * hr[q] - ati * hi[q] + br;
      float ni = atr * hi[q] + ati * hr[q] + bi;
      hr[q] = nr; hi[q] = ni;
      float qr = atr * pr[q] - ati * pi[q];
      float qi = atr * pi[q] + ati * pr[q];
      pr[q] = qr; pi[q] = qi;
    }
  }
  #pragma unroll
  for (int q = 0; q < 2; ++q) {
    int s = lane + q * 32;
    int o = ((c * DINNER + d) * DSTATE + s) * 2;
    chp[o] = pr[q]; chp[o + 1] = pi[q];
    chh[o] = hr[q]; chh[o + 1] = hi[q];
  }
}

// -------------- scan pass 2: stitch chunk boundaries (serial over 16) ------
__global__ void k_scan2(const float* __restrict__ chp, const float* __restrict__ chh,
                        float* __restrict__ hstart) {
  int idx = blockIdx.x * blockDim.x + threadIdx.x;   // d*64 + s  (8192)
  float hr = 0.f, hi = 0.f;
  for (int c = 0; c < NCH; ++c) {
    int o = (c * (DINNER * DSTATE) + idx) * 2;
    hstart[o] = hr; hstart[o + 1] = hi;
    float pr = chp[o], pi = chp[o + 1];
    float br = chh[o], bi = chh[o + 1];
    float nr = pr * hr - pi * hi + br;
    float ni = pr * hi + pi * hr + bi;
    hr = nr; hi = ni;
  }
}

// -------------- scan pass 3: replay chunk, emit y via wave reduction -------
__global__ void k_scan3(const float* __restrict__ dtb, const float* __restrict__ xc,
                        const float* __restrict__ xdbl,
                        const float* __restrict__ alog, const float* __restrict__ aimg,
                        const float* __restrict__ hstart, float* __restrict__ y) {
  int gwave = (blockIdx.x * blockDim.x + threadIdx.x) >> 5;
  int lane  = threadIdx.x & 31;
  int c = gwave >> 7, d = gwave & 127;
  float ar[2], ai[2], hr[2], hi[2];
  #pragma unroll
  for (int q = 0; q < 2; ++q) {
    int s = lane + q * 32;
    ar[q] = -__expf(alog[d * DSTATE + s]);
    ai[q] = aimg[d * DSTATE + s];
    int o = ((c * DINNER + d) * DSTATE + s) * 2;
    hr[q] = hstart[o]; hi[q] = hstart[o + 1];
  }
  int tbeg = c * CHUNK;
  for (int t = tbeg; t < tbeg + CHUNK; ++t) {
    float dv = dtb[t * DINNER + d];
    float u  = dv * xc[t * DINNER + d];
    const float* row = xdbl + t * NXROW;
    float yv = 0.f;
    #pragma unroll
    for (int q = 0; q < 2; ++q) {
      int s = lane + q * 32;
      float ea  = __expf(dv * ar[q]);
      float th  = dv * ai[q];
      float atr = ea * __cosf(th), ati = ea * __sinf(th);
      float br = u * row[4 + s], bi = u * row[68 + s];
      float nr = atr * hr[q] - ati * hi[q] + br;
      float ni = atr * hi[q] + ati * hr[q] + bi;
      hr[q] = nr; hi[q] = ni;
      yv += nr * row[132 + s] - ni * row[196 + s];   // Re(h * C)
    }
    #pragma unroll
    for (int off = 16; off; off >>= 1) yv += __shfl_xor(yv, off);
    if (lane == 0) y[t * DINNER + d] = yv;
  }
}

// ----------------------- gate: (y + D*xc) * silu(z) ------------------------
__global__ void k_gate(const float* __restrict__ y, const float* __restrict__ xz,
                       const float* __restrict__ xc32, const float* __restrict__ Dp,
                       _Float16* __restrict__ y16) {
  int idx = blockIdx.x * blockDim.x + threadIdx.x;   // l*128 + d
  int l = idx >> 7, d = idx & 127;
  float zv = xz[l * (2 * DINNER) + DINNER + d];
  float g  = zv / (1.f + __expf(-zv));
  y16[idx] = (_Float16)((y[idx] + Dp[d] * xc32[idx]) * g);
}

// ----------------------------- final add ----------------------------------
__global__ void k_final(const float* __restrict__ hidden, const float* __restrict__ residual,
                        float* __restrict__ out) {
  int i = blockIdx.x * blockDim.x + threadIdx.x;
  out[i] = hidden[i] + residual[i];
}

// ---------------------------------------------------------------------------
extern "C" void kernel_launch(void* const* d_in, const int* in_sizes, int n_in,
                              void* d_out, int out_size, void* d_ws, size_t ws_size,
                              hipStream_t stream) {
  (void)in_sizes; (void)n_in; (void)out_size; (void)ws_size;
  const float* x          = (const float*)d_in[0];
  const float* norm_w     = (const float*)d_in[1];
  const float* in_proj_w  = (const float*)d_in[2];
  const float* conv_w     = (const float*)d_in[3];
  const float* conv_b     = (const float*)d_in[4];
  const float* x_proj_w   = (const float*)d_in[5];
  const float* dt_proj_w  = (const float*)d_in[6];
  const float* dt_bias    = (const float*)d_in[7];
  const float* A_log      = (const float*)d_in[8];
  const float* A_imag     = (const float*)d_in[9];
  const float* Dskip      = (const float*)d_in[10];
  const float* out_proj_w = (const float*)d_in[11];

  char* ws = (char*)d_ws;
  size_t off = 0;
  auto alloc = [&](size_t bytes) -> char* {
    char* p = ws + off;
    off += (bytes + 255) & ~(size_t)255;
    return p;
  };
  float*     residual = (float*)    alloc(LSEQ * DMODEL * 4);
  float*     hidden   = (float*)    alloc(LSEQ * DMODEL * 4);
  _Float16*  h16      = (_Float16*) alloc(LSEQ * DMODEL * 2);
  float*     xz       = (float*)    alloc(LSEQ * 2 * DINNER * 4);
  float*     xc32     = (float*)    alloc(LSEQ * DINNER * 4);
  _Float16*  xc16     = (_Float16*) alloc(LSEQ * DINNER * 2);
  float*     xdbl     = (float*)    alloc(LSEQ * NXROW * 4);
  float*     dtb      = (float*)    alloc(LSEQ * DINNER * 4);
  float*     ybuf     = (float*)    alloc(LSEQ * DINNER * 4);
  _Float16*  y16      = (_Float16*) alloc(LSEQ * DINNER * 2);
  float*     chp      = (float*)    alloc((size_t)NCH * DINNER * DSTATE * 2 * 4);
  float*     chh      = (float*)    alloc((size_t)NCH * DINNER * DSTATE * 2 * 4);
  float*     hstart   = (float*)    alloc((size_t)NCH * DINNER * DSTATE * 2 * 4);
  _Float16*  ipw16    = (_Float16*) alloc((size_t)NLAYER * 2 * DINNER * DMODEL * 2);
  _Float16*  xpw16    = (_Float16*) alloc((size_t)NLAYER * NXROW * DINNER * 2);
  _Float16*  opw16    = (_Float16*) alloc((size_t)NLAYER * DMODEL * DINNER * 2);

  // one-time (per launch) weight conversion to f16
  k_cvt<<<(NLAYER * 2 * DINNER * DMODEL) / 256, 256, 0, stream>>>(in_proj_w,  ipw16);
  k_cvt<<<(NLAYER * NXROW * DINNER)      / 256, 256, 0, stream>>>(x_proj_w,   xpw16);
  k_cvt<<<(NLAYER * DMODEL * DINNER)     / 256, 256, 0, stream>>>(out_proj_w, opw16);

  k_init<<<(LSEQ * DMODEL) / 256, 256, 0, stream>>>(x, hidden, residual);

  for (int layer = 0; layer < NLAYER; ++layer) {
    const float*    nw  = norm_w    + layer * DMODEL;
    const _Float16* ipw = ipw16     + layer * 2 * DINNER * DMODEL;
    const float*    cw  = conv_w    + layer * DINNER * 4;
    const float*    cbp = conv_b    + layer * DINNER;
    const _Float16* xpw = xpw16     + layer * NXROW * DINNER;
    const float*    dpw = dt_proj_w + layer * DINNER * 4;
    const float*    dbp = dt_bias   + layer * DINNER;
    const float*    alp = A_log     + layer * DINNER * DSTATE;
    const float*    aip = A_imag    + layer * DINNER * DSTATE;
    const float*    dsp = Dskip     + layer * DINNER;
    const _Float16* opw = opw16     + layer * DMODEL * DINNER;

    k_norm<<<LSEQ, DMODEL, 0, stream>>>(hidden, residual, nw, h16);

    {   // xz = h @ ipw^T : M=2048 N=256 K=64 -> 128x8 tile-pairs, 8 waves/blk
      int npairs = (2 * DINNER) / 32, waves = (LSEQ / 16) * npairs;
      k_gemm<<<waves * 32 / 256, 256, 0, stream>>>(h16, DMODEL, ipw, DMODEL,
                                                   xz, 2 * DINNER, LSEQ, 2 * DINNER, DMODEL, npairs);
    }
    k_conv<<<(LSEQ * DINNER) / 256, 256, 0, stream>>>(xz, cw, cbp, xc32, xc16);
    {   // xdbl = xc @ xpw^T : M=2048 N=260 K=128 (ragged N: clamp + store guard)
      int npairs = (NXROW + 31) / 32, waves = (LSEQ / 16) * npairs;
      k_gemm<<<waves * 32 / 256, 256, 0, stream>>>(xc16, DINNER, xpw, DINNER,
                                                   xdbl, NXROW, LSEQ, NXROW, DINNER, npairs);
    }
    k_dt  <<<(LSEQ * DINNER) / 256, 256, 0, stream>>>(xdbl, dpw, dbp, dtb);
    k_scan1<<<(NCH * DINNER) / 8, 256, 0, stream>>>(dtb, xc32, xdbl, alp, aip, chp, chh);
    k_scan2<<<(DINNER * DSTATE) / 256, 256, 0, stream>>>(chp, chh, hstart);
    k_scan3<<<(NCH * DINNER) / 8, 256, 0, stream>>>(dtb, xc32, xdbl, alp, aip, hstart, ybuf);
    k_gate<<<(LSEQ * DINNER) / 256, 256, 0, stream>>>(ybuf, xz, xc32, dsp, y16);
    {   // hidden = y @ opw^T : M=2048 N=64 K=128
      int npairs = DMODEL / 32, waves = (LSEQ / 16) * npairs;
      k_gemm<<<waves * 32 / 256, 256, 0, stream>>>(y16, DINNER, opw, DINNER,
                                                   hidden, DMODEL, LSEQ, DMODEL, DINNER, npairs);
    }
  }
  k_final<<<(LSEQ * DMODEL) / 256, 256, 0, stream>>>(hidden, residual, (float*)d_out);
}
